// Model_37666863186261
// MI455X (gfx1250) — matrix operand order repaired and checked by reading, not compiled, not run on hardware
//
#include <hip/hip_runtime.h>
#include <hip/hip_bf16.h>

// ---------------- model constants ----------------
#define BB 8
#define LEN 1024
#define LDEC 768
#define PRED 384
#define DM 512
#define NH 8
#define DKH 64
#define DFF 2048
#define CIN 7
#define MARKD 4
#define NTOP 35
#define SAMPLEK 35
#define NEGINF -3.4e38f

// ---------------- WMMA types ----------------
typedef __attribute__((ext_vector_type(16))) __bf16 v16bf;
typedef __attribute__((ext_vector_type(8)))  float  v8f;

__device__ __forceinline__ unsigned short f2bf(float f) {
    unsigned u = __float_as_uint(f);
    unsigned r = (u + 0x7FFFu + ((u >> 16) & 1u)) >> 16;   // round-to-nearest-even
    return (unsigned short)r;
}
__device__ __forceinline__ unsigned pack_bf2(float lo, float hi) {
    return (unsigned)f2bf(lo) | ((unsigned)f2bf(hi) << 16);
}

#define LDT 40   // LDS tile row stride in ushorts (80B rows: 16B aligned, conflict-free)

// A fragment: 16x32 bf16.  lanes 0-15: row=base+lane, K 0..15;
// lanes 16-31: same rows, K 16..31 (matches B fragment k order).
__device__ __forceinline__ v16bf load_a_frag(const unsigned short As[64][LDT], int rbase, int lane) {
    union { unsigned u[8]; v16bf v; } f;
    int row = rbase + (lane & 15);
    int kb  = (lane >> 4) << 4;            // 0 or 16
#pragma unroll
    for (int i = 0; i < 4; i++) f.u[i]     = *(const unsigned*)&As[row][kb + 2 * i];
#pragma unroll
    for (int i = 0; i < 4; i++) f.u[4 + i] = *(const unsigned*)&As[row][kb + 8 + 2 * i];
    return f.v;
}

// B fragment: 32x16 bf16 (Bs stored [n][k]).  lanes 0-15: col=base+lane, K 0..15;
// lanes 16-31: K 16..31.
__device__ __forceinline__ v16bf load_b_frag(const unsigned short Bs[64][LDT], int cbase, int lane) {
    union { unsigned u[8]; v16bf v; } f;
    int col = cbase + (lane & 15);
    int kb  = (lane >> 4) << 4;            // 0 or 16
#pragma unroll
    for (int i = 0; i < 8; i++) f.u[i] = *(const unsigned*)&Bs[col][kb + 2 * i];
    return f.v;
}

// ---------------- operand conversion passes (done ONCE per GEMM operand) ----------------
// fp32 -> bf16, 8 elems/thread, packed stores. n % 8 == 0 at all call sites.
__global__ void a2bf_kernel(const float* __restrict__ x, unsigned short* __restrict__ y, int n) {
    const int i = (blockIdx.x * blockDim.x + threadIdx.x) * 8;
    if (i >= n) return;
    const float4 a = *(const float4*)(x + i);
    const float4 b = *(const float4*)(x + i + 4);
    uint4 o;
    o.x = pack_bf2(a.x, a.y);
    o.y = pack_bf2(a.z, a.w);
    o.z = pack_bf2(b.x, b.y);
    o.w = pack_bf2(b.z, b.w);
    *(uint4*)(y + i) = o;
}

// fp32 W[KxN] -> bf16 Wt[NxK] (transpose via 32x32 LDS tile). K,N multiples of 32.
__global__ __launch_bounds__(256) void wconv_kernel(const float* __restrict__ W,
                                                    unsigned short* __restrict__ Wt,
                                                    int K, int N) {
    __shared__ float tile[32][33];
    const int nb = blockIdx.x * 32, kb = blockIdx.y * 32;
    const int tx = threadIdx.x & 31, ty = threadIdx.x >> 5;   // 32 x 8
#pragma unroll
    for (int i = ty; i < 32; i += 8) tile[i][tx] = W[(size_t)(kb + i) * N + nb + tx];
    __syncthreads();
#pragma unroll
    for (int i = ty; i < 32; i += 8)
        Wt[(size_t)(nb + i) * K + kb + tx] = f2bf(tile[tx][i]);
}

// ---------------- bf16 WMMA GEMM: out = epilogue(A[MxK] @ W[KxN] + bias) ----------------
// A: bf16 row-major MxK.  Wt: bf16 N x K (pre-transposed).  bias/res/out: fp32.
// EPI: 0 = bias, 1 = bias+exact GELU, 2 = bias + residual add.
// M % 64 == 0, N % 64 == 0, K % 32 == 0 at all call sites.
template <int EPI>
__global__ __launch_bounds__(128) void gemm_bf16_kernel(
        const unsigned short* __restrict__ A, const unsigned short* __restrict__ Wt,
        const float* __restrict__ bias, const float* __restrict__ res,
        float* __restrict__ out, int M, int N, int K) {
    __shared__ unsigned short As[64][LDT];
    __shared__ unsigned short Bs[64][LDT];

    const int t    = threadIdx.x;
    const int lane = t & 31;
    const int wave = t >> 5;
    const int wm   = wave >> 1;    // 0..1 : 32-row slab
    const int wn   = wave & 1;     // 0..1 : 32-col slab
    const int m0   = blockIdx.y * 64;
    const int n0   = blockIdx.x * 64;

    // staging coords: each thread copies one 8-element (16B) chunk of A and of B
    const int sr = t >> 1;               // 0..63
    const int sk = (t & 1) * 16;         // 0 or 16

    const v8f vzero = {0.f, 0.f, 0.f, 0.f, 0.f, 0.f, 0.f, 0.f};
    v8f acc[2][2] = {{vzero, vzero}, {vzero, vzero}};

    for (int k0 = 0; k0 < K; k0 += 32) {
        // b128 copy global -> LDS, no conversion, no transpose (both operands contiguous)
        *(uint4*)&As[sr][sk] = *(const uint4*)(A  + (size_t)(m0 + sr) * K + k0 + sk);
        *(uint4*)&Bs[sr][sk] = *(const uint4*)(Wt + (size_t)(n0 + sr) * K + k0 + sk);
        __syncthreads();

        v16bf a0 = load_a_frag(As, wm * 32, lane);
        v16bf a1 = load_a_frag(As, wm * 32 + 16, lane);
        v16bf b0 = load_b_frag(Bs, wn * 32, lane);
        v16bf b1 = load_b_frag(Bs, wn * 32 + 16, lane);

        acc[0][0] = __builtin_amdgcn_wmma_f32_16x16x32_bf16(false, a0, false, b0, (short)0, acc[0][0], false, false);
        acc[0][1] = __builtin_amdgcn_wmma_f32_16x16x32_bf16(false, a0, false, b1, (short)0, acc[0][1], false, false);
        acc[1][0] = __builtin_amdgcn_wmma_f32_16x16x32_bf16(false, a1, false, b0, (short)0, acc[1][0], false, false);
        acc[1][1] = __builtin_amdgcn_wmma_f32_16x16x32_bf16(false, a1, false, b1, (short)0, acc[1][1], false, false);
        __syncthreads();
    }

    // C/D layout: lanes 0-15 -> M=R+v, lanes 16-31 -> M=R+8+v; N = C + (lane&15)
#pragma unroll
    for (int i = 0; i < 2; i++) {
#pragma unroll
        for (int j = 0; j < 2; j++) {
            const int Rb = m0 + wm * 32 + i * 16 + ((lane >> 4) << 3);
            const int Cc = n0 + wn * 32 + j * 16 + (lane & 15);
            const float bv = bias[Cc];
#pragma unroll
            for (int v = 0; v < 8; v++) {
                const int r = Rb + v;
                float val = acc[i][j][v] + bv;
                if (EPI == 1) val = 0.5f * val * (1.f + erff(val * 0.70710678118654752f));
                if (EPI == 2) val += res[(size_t)r * N + Cc];
                out[(size_t)r * N + Cc] = val;
            }
        }
    }
}

// ---------------- layernorm: one wave32 per 512-wide row ----------------
__global__ void layernorm_kernel(const float* __restrict__ x, const float* __restrict__ g,
                                 const float* __restrict__ b, float* __restrict__ out, int rows) {
    const int row  = blockIdx.x;
    const int lane = threadIdx.x;
    if (row >= rows) return;
    const float* xr = x + (size_t)row * DM;
    float v[16];
    float s = 0.f;
#pragma unroll
    for (int i = 0; i < 16; i++) { v[i] = xr[lane + i * 32]; s += v[i]; }
#pragma unroll
    for (int o = 16; o > 0; o >>= 1) s += __shfl_xor(s, o, 32);
    const float mu = s * (1.f / DM);
    float vs = 0.f;
#pragma unroll
    for (int i = 0; i < 16; i++) { const float d = v[i] - mu; vs += d * d; }
#pragma unroll
    for (int o = 16; o > 0; o >>= 1) vs += __shfl_xor(vs, o, 32);
    const float rinv = rsqrtf(vs * (1.f / DM) + 1e-5f);
    float* orow = out + (size_t)row * DM;
#pragma unroll
    for (int i = 0; i < 16; i++) {
        const int c = lane + i * 32;
        orow[c] = g[c] * (v[i] - mu) * rinv + b[c];
    }
}

// ---------------- data embedding: circular conv3 + positional + temporal ----------------
__global__ void embed_kernel(const float* __restrict__ x, const float* __restrict__ xm,
                             const float* __restrict__ cw, const float* __restrict__ tw,
                             const float* __restrict__ tb, float* __restrict__ out, int L) {
    const int row = blockIdx.x;           // b*L + l
    const int b = row / L, l = row % L;
    for (int d = threadIdx.x; d < DM; d += blockDim.x) {
        float s = tb[d];
#pragma unroll
        for (int tt = 0; tt < 3; tt++) {
            const int ls = (l + tt - 1 + L) % L;
            const float* xr = x + (size_t)(b * L + ls) * CIN;
#pragma unroll
            for (int c = 0; c < CIN; c++) s += xr[c] * cw[(tt * CIN + c) * DM + d];
        }
        const float* xmr = xm + (size_t)row * MARKD;
#pragma unroll
        for (int m = 0; m < MARKD; m++) s += xmr[m] * tw[m * DM + d];
        const int i2 = d & ~1;
        const float freq = __expf(-(float)i2 * (9.210340371976184f / (float)DM)); // ln(1e4)
        const float ang  = (float)l * freq;
        s += (d & 1) ? __cosf(ang) : __sinf(ang);
        out[(size_t)row * DM + d] = s;
    }
}

// ---------------- prob-sparse: sampled sparsity score M per query (wave per query) ------
__global__ void mscore_kernel(const float* __restrict__ Q, const float* __restrict__ Kc,
                              float* __restrict__ Mout, int Lq, int Lk, unsigned seed) {
    const int gwave = (blockIdx.x * blockDim.x + threadIdx.x) >> 5;
    const int lane  = threadIdx.x & 31;
    const int l  = gwave % Lq;
    const int bh = gwave / Lq;
    const int h  = bh % NH, b = bh / NH;
    const float* q = Q + (size_t)(b * Lq + l) * DM + h * DKH;
    const float q0 = q[lane], q1 = q[lane + 32];
    float mx = NEGINF, sm = 0.f;
    for (int s = 0; s < SAMPLEK; s++) {
        unsigned hh = (unsigned)(l * SAMPLEK + s) * 2654435761u ^ seed;
        hh ^= hh >> 13; hh *= 0x9E3779B1u; hh ^= hh >> 16;
        const int idx = (int)(hh % (unsigned)Lk);
        const float* kp = Kc + (size_t)(b * Lk + idx) * DM + h * DKH;
        float p = q0 * kp[lane] + q1 * kp[lane + 32];
#pragma unroll
        for (int o = 16; o > 0; o >>= 1) p += __shfl_xor(p, o, 32);
        mx = fmaxf(mx, p);
        sm += p;
    }
    if (lane == 0) Mout[gwave] = mx - sm / (float)Lk;
}

// ---------------- iterative top-k (k=35) per (b,h) ----------------
__global__ void topk_kernel(const float* __restrict__ M, int* __restrict__ idxt, int Lq) {
    __shared__ float vals[1024];
    __shared__ float rv[256];
    __shared__ int   ri[256];
    const int bh = blockIdx.x, t = threadIdx.x;
    const float* m = M + (size_t)bh * Lq;
    for (int i = t; i < Lq; i += 256) vals[i] = m[i];
    __syncthreads();
    for (int u = 0; u < NTOP; u++) {
        float best = NEGINF; int bi = 0;
        for (int i = t; i < Lq; i += 256)
            if (vals[i] > best) { best = vals[i]; bi = i; }
        rv[t] = best; ri[t] = bi;
        __syncthreads();
        for (int o = 128; o > 0; o >>= 1) {
            if (t < o && rv[t + o] > rv[t]) { rv[t] = rv[t + o]; ri[t] = ri[t + o]; }
            __syncthreads();
        }
        if (t == 0) { idxt[bh * NTOP + u] = ri[0]; vals[ri[0]] = NEGINF; }
        __syncthreads();
    }
}

// ---------------- context init: V column means (non-causal) ----------------
__global__ void vmean_kernel(const float* __restrict__ V, float* __restrict__ vm, int Lk) {
    const int i = blockIdx.x * blockDim.x + threadIdx.x;   // B*DM threads
    const int b = i / DM, c = i % DM;
    float s = 0.f;
    for (int l = 0; l < Lk; l++) s += V[(size_t)(b * Lk + l) * DM + c];
    vm[i] = s / (float)Lk;
}

__global__ void ctxfill_kernel(const float* __restrict__ vm, float* __restrict__ ctx, int Lq) {
    const int i = blockIdx.x * blockDim.x + threadIdx.x;   // B*Lq*DM
    const int c = i % DM;
    const int b = i / (Lq * DM);
    ctx[i] = vm[b * DM + c];
}

// ---------------- context init: cumsum of V along L (causal) ----------------
__global__ void cumsum_kernel(const float* __restrict__ V, float* __restrict__ ctx, int L) {
    const int i = blockIdx.x * blockDim.x + threadIdx.x;   // B*DM threads
    const int b = i / DM, c = i % DM;
    float s = 0.f;
    for (int l = 0; l < L; l++) {
        s += V[(size_t)(b * L + l) * DM + c];
        ctx[(size_t)(b * L + l) * DM + c] = s;
    }
}

// ---------------- full attention row for selected queries, scatter into ctx ----------------
__global__ __launch_bounds__(256) void topattn_kernel(
        const float* __restrict__ Q, const float* __restrict__ Kc, const float* __restrict__ Vc,
        const int* __restrict__ idxt, float* __restrict__ ctx, int Lq, int Lk, int causal) {
    __shared__ float qs[DKH];
    __shared__ float sc[1024];
    __shared__ float red[256];
    const int u  = blockIdx.x % NTOP;
    const int bh = blockIdx.x / NTOP;
    const int h = bh % NH, b = bh / NH;
    const int t = threadIdx.x;
    const int lstar = idxt[bh * NTOP + u];
    const float* q = Q + (size_t)(b * Lq + lstar) * DM + h * DKH;
    if (t < DKH) qs[t] = q[t];
    __syncthreads();

    for (int l = t; l < Lk; l += 256) {
        const float* kp = Kc + (size_t)(b * Lk + l) * DM + h * DKH;
        float s = 0.f;
#pragma unroll
        for (int d = 0; d < DKH; d++) s += qs[d] * kp[d];
        s *= 0.125f;                                   // 1/sqrt(64)
        if (causal && l > lstar) s = -1e9f;
        sc[l] = s;
    }
    __syncthreads();

    float mx = NEGINF;
    for (int l = t; l < Lk; l += 256) mx = fmaxf(mx, sc[l]);
    red[t] = mx; __syncthreads();
    for (int o = 128; o > 0; o >>= 1) { if (t < o) red[t] = fmaxf(red[t], red[t + o]); __syncthreads(); }
    mx = red[0]; __syncthreads();

    float sm = 0.f;
    for (int l = t; l < Lk; l += 256) { const float e = __expf(sc[l] - mx); sc[l] = e; sm += e; }
    red[t] = sm; __syncthreads();
    for (int o = 128; o > 0; o >>= 1) { if (t < o) red[t] += red[t + o]; __syncthreads(); }
    const float inv = 1.f / red[0]; __syncthreads();

    // weighted sum over V: thread -> (chunk, d)
    const int d = t & 63, ch = t >> 6;
    float acc = 0.f;
    for (int l = ch; l < Lk; l += 4)
        acc += sc[l] * Vc[(size_t)(b * Lk + l) * DM + h * DKH + d];
    red[t] = acc; __syncthreads();
    if (ch == 0) {
        const float r = red[d] + red[64 + d] + red[128 + d] + red[192 + d];
        ctx[(size_t)(b * Lq + lstar) * DM + h * DKH + d] = r * inv;
    }
}

// ---------------- conv-distill helpers ----------------
// im2col emits bf16 directly (only ever consumed as a GEMM A-operand)
__global__ void im2col_bf_kernel(const float* __restrict__ x, unsigned short* __restrict__ out,
                                 int L, int total) {
    const int i = blockIdx.x * blockDim.x + threadIdx.x;
    if (i >= total) return;
    const int col = i % (3 * DM);
    const int row = i / (3 * DM);
    const int b = row / L, l = row % L;
    const int tt = col / DM, c = col % DM;
    const int ls = (l + tt - 1 + L) % L;
    out[i] = f2bf(x[(size_t)(b * L + ls) * DM + c]);
}

__global__ void bnstats_kernel(const float* __restrict__ y, float* __restrict__ mean,
                               float* __restrict__ var, int rows) {
    const int c = blockIdx.x * blockDim.x + threadIdx.x;
    if (c >= DM) return;
    float s = 0.f, s2 = 0.f;
    for (int r = 0; r < rows; r++) { const float v = y[(size_t)r * DM + c]; s += v; s2 += v * v; }
    const float mu = s / (float)rows;
    mean[c] = mu;
    var[c] = s2 / (float)rows - mu * mu;
}

__global__ void bn_elu_kernel(const float* __restrict__ y, const float* __restrict__ mean,
                              const float* __restrict__ var, const float* __restrict__ g,
                              const float* __restrict__ bb, float* __restrict__ out, int n) {
    const int i = blockIdx.x * blockDim.x + threadIdx.x;
    if (i >= n) return;
    const int c = i % DM;
    const float v = (y[i] - mean[c]) * rsqrtf(var[c] + 1e-5f) * g[c] + bb[c];
    out[i] = v > 0.f ? v : (__expf(v) - 1.f);
}

__global__ void maxpool_kernel(const float* __restrict__ x, float* __restrict__ out, int Lin, int n) {
    const int i = blockIdx.x * blockDim.x + threadIdx.x;
    if (i >= n) return;
    const int Lo = Lin >> 1;
    const int c = i % DM;
    const int rest = i / DM;
    const int lo = rest % Lo, b = rest / Lo;
    float m = NEGINF;
#pragma unroll
    for (int dl = -1; dl <= 1; dl++) {
        const int li = 2 * lo + dl;
        if (li >= 0 && li < Lin) m = fmaxf(m, x[(size_t)(b * Lin + li) * DM + c]);
    }
    out[i] = m;
}

// ---------------- final projection 512 -> 7, last PRED rows ----------------
__global__ void proj_kernel(const float* __restrict__ x, const float* __restrict__ w,
                            const float* __restrict__ bb, float* __restrict__ out) {
    const int i = blockIdx.x * blockDim.x + threadIdx.x;
    if (i >= BB * PRED * 7) return;
    const int c = i % 7;
    const int rest = i / 7;
    const int lo = rest % PRED, b = rest / PRED;
    const int ls = lo + (LDEC - PRED);
    const float* xr = x + (size_t)(b * LDEC + ls) * DM;
    float s = bb[c];
    for (int d = 0; d < DM; d++) s += xr[d] * w[d * 7 + c];
    out[i] = s;
}

// =======================================================================================
extern "C" void kernel_launch(void* const* d_in, const int* in_sizes, int n_in,
                              void* d_out, int out_size, void* d_ws, size_t ws_size,
                              hipStream_t stream) {
    (void)in_sizes; (void)n_in; (void)out_size; (void)ws_size;

    // Input order: x_enc, x_mark_enc, x_dec, x_mark_dec, then params pytree-flattened
    // (dict keys sorted): dec_emb{conv_w,temp_b,temp_w}, dec_layers[0]{cross{bk,bo,bq,bv,
    // wk,wo,wq,wv}, ff1_b, ff1_w, ff2_b, ff2_w, ln1{b,g}, ln2{b,g}, ln3{b,g},
    // self{bk,bo,bq,bv,wk,wo,wq,wv}}, dec_norm{b,g}, enc_convs[0]{b,bn_b,bn_g,w},
    // enc_emb{conv_w,temp_b,temp_w}, enc_layers[i]{attn{..8..},ff1_b,ff1_w,ff2_b,ff2_w,
    // ln1{b,g},ln2{b,g}}, enc_norm{b,g}, proj_b, proj_w.
    auto P = [&](int i) -> const float* { return (const float*)d_in[i]; };

    float* W = (float*)d_ws;
    size_t off = 0;
    auto alloc = [&](size_t n) -> float* { float* p = W + off; off += (n + 63) & ~(size_t)63; return p; };

    const size_t TMAX = (size_t)BB * LEN;          // 8192 tokens max
    float* bufX   = alloc(TMAX * DM);
    float* bufX2  = alloc(TMAX * DM);
    float* bufX3  = alloc(TMAX * DM);
    float* bufT   = alloc(TMAX * DM);
    float* bufQ   = alloc(TMAX * DM);
    float* bufK   = alloc(TMAX * DM);
    float* bufV   = alloc(TMAX * DM);
    float* bufCtx = alloc(TMAX * DM);
    float* bufFF  = alloc(TMAX * DFF);
    float* encOut = alloc((size_t)BB * 512 * DM);
    float* bufM   = alloc((size_t)BB * NH * LEN);
    float* vmeanb = alloc((size_t)BB * DM);
    float* meanb  = alloc(DM);
    float* varb   = alloc(DM);
    int*   bufIdx = (int*)alloc(BB * NH * NTOP);
    // bf16 operand staging (ushorts; sized in float units = n/2)
    unsigned short* bfA = (unsigned short*)alloc(TMAX * DFF / 2);   // largest A: 8192x2048
    unsigned short* bfB = (unsigned short*)alloc(TMAX * DM / 2);    // kv operand (cross attn)
    unsigned short* bfW = (unsigned short*)alloc((size_t)DM * DFF / 2); // largest weight

    auto toBF = [&](const float* src, unsigned short* dst, size_t n) {
        a2bf_kernel<<<((unsigned)(n / 8) + 255) / 256, 256, 0, stream>>>(src, dst, (int)n);
    };
    auto wToBF = [&](const float* w, int K, int N) {   // -> bfW as [N][K]
        dim3 g(N / 32, K / 32);
        wconv_kernel<<<g, 256, 0, stream>>>(w, bfW, K, N);
    };
    auto gemmB = [&](const unsigned short* A, const float* bias, float* out,
                     int M, int N, int K) {
        dim3 g(N / 64, M / 64);
        gemm_bf16_kernel<0><<<g, 128, 0, stream>>>(A, bfW, bias, nullptr, out, M, N, K);
    };
    auto gemmG = [&](const unsigned short* A, const float* bias, float* out,
                     int M, int N, int K) {
        dim3 g(N / 64, M / 64);
        gemm_bf16_kernel<1><<<g, 128, 0, stream>>>(A, bfW, bias, nullptr, out, M, N, K);
    };
    auto gemmR = [&](const unsigned short* A, const float* bias, const float* res,
                     float* out, int M, int N, int K) {
        dim3 g(N / 64, M / 64);
        gemm_bf16_kernel<2><<<g, 128, 0, stream>>>(A, bfW, bias, res, out, M, N, K);
    };
    auto ln = [&](const float* x, const float* g, const float* b, float* out, int rows) {
        layernorm_kernel<<<rows, 32, 0, stream>>>(x, g, b, out, rows);
    };

    // prob-sparse attention block: out = xq + (ctx @ wo + bo)
    auto attn_block = [&](const float* xq, const float* xkv, int Lq, int Lk,
                          const float* wq, const float* bq, const float* wk, const float* bk,
                          const float* wv, const float* bv, const float* wo, const float* bo,
                          int causal, unsigned seed, float* out) {
        toBF(xq, bfA, (size_t)BB * Lq * DM);
        const unsigned short* bkv = bfA;
        if (xkv != xq) { toBF(xkv, bfB, (size_t)BB * Lk * DM); bkv = bfB; }
        wToBF(wq, DM, DM); gemmB(bfA, bq, bufQ, BB * Lq, DM, DM);
        wToBF(wk, DM, DM); gemmB(bkv, bk, bufK, BB * Lk, DM, DM);
        wToBF(wv, DM, DM); gemmB(bkv, bv, bufV, BB * Lk, DM, DM);
        mscore_kernel<<<(BB * NH * Lq) / 8, 256, 0, stream>>>(bufQ, bufK, bufM, Lq, Lk, seed);
        topk_kernel<<<BB * NH, 256, 0, stream>>>(bufM, bufIdx, Lq);
        if (causal) {
            cumsum_kernel<<<(BB * DM) / 256, 256, 0, stream>>>(bufV, bufCtx, Lq);
        } else {
            vmean_kernel<<<(BB * DM) / 256, 256, 0, stream>>>(bufV, vmeanb, Lk);
            ctxfill_kernel<<<(BB * Lq * DM) / 256, 256, 0, stream>>>(vmeanb, bufCtx, Lq);
        }
        topattn_kernel<<<BB * NH * NTOP, 256, 0, stream>>>(bufQ, bufK, bufV, bufIdx, bufCtx,
                                                           Lq, Lk, causal);
        toBF(bufCtx, bfA, (size_t)BB * Lq * DM);
        wToBF(wo, DM, DM);
        gemmR(bfA, bo, xq, out, BB * Lq, DM, DM);
    };

    // encoder layer at param base pb: bk+0 bo+1 bq+2 bv+3 wk+4 wo+5 wq+6 wv+7
    //                                 ff1_b+8 ff1_w+9 ff2_b+10 ff2_w+11 ln1{b12,g13} ln2{b14,g15}
    auto enc_layer = [&](float* x, int L, int pb, unsigned seed) {
        attn_block(x, x, L, L, P(pb + 6), P(pb + 2), P(pb + 4), P(pb + 0),
                   P(pb + 7), P(pb + 3), P(pb + 5), P(pb + 1), 0, seed, bufT);
        ln(bufT, P(pb + 13), P(pb + 12), bufX2, BB * L);
        toBF(bufX2, bfA, (size_t)BB * L * DM);
        wToBF(P(pb + 9), DM, DFF);
        gemmG(bfA, P(pb + 8), bufFF, BB * L, DFF, DM);
        toBF(bufFF, bfA, (size_t)BB * L * DFF);
        wToBF(P(pb + 11), DFF, DM);
        gemmR(bfA, P(pb + 10), bufX2, bufT, BB * L, DM, DFF);
        ln(bufT, P(pb + 15), P(pb + 14), x, BB * L);
    };

    // ---------------- encoder ----------------
    embed_kernel<<<BB * LEN, 256, 0, stream>>>(P(0), P(1), P(39), P(41), P(40), bufX, LEN);
    enc_layer(bufX, LEN, 42, 0xA341316Cu);

    // distilling conv: circular conv3 as im2col (K=1536) GEMM, then BN + ELU + maxpool/2
    {
        const int total = BB * LEN * 3 * DM;
        im2col_bf_kernel<<<(total + 255) / 256, 256, 0, stream>>>(bufX, bfA, LEN, total);
        wToBF(P(38), 3 * DM, DM);
        gemmB(bfA, P(35), bufT, BB * LEN, DM, 3 * DM);
        bnstats_kernel<<<2, 256, 0, stream>>>(bufT, meanb, varb, BB * LEN);
        const int n = BB * LEN * DM;
        bn_elu_kernel<<<(n + 255) / 256, 256, 0, stream>>>(bufT, meanb, varb, P(37), P(36), bufX2, n);
        const int np = BB * (LEN / 2) * DM;
        maxpool_kernel<<<(np + 255) / 256, 256, 0, stream>>>(bufX2, bufX, LEN, np);
    }

    enc_layer(bufX, 512, 58, 0x8F2D1E4Bu);
    ln(bufX, P(75), P(74), encOut, BB * 512);

    // ---------------- decoder ----------------
    embed_kernel<<<BB * LDEC, 256, 0, stream>>>(P(2), P(3), P(4), P(6), P(5), bufX, LDEC);
    // self attention (causal prob-sparse): self{bk25,bo26,bq27,bv28,wk29,wo30,wq31,wv32}
    attn_block(bufX, bufX, LDEC, LDEC, P(31), P(27), P(29), P(25),
               P(32), P(28), P(30), P(26), 1, 0x13C6EF37u, bufT);
    ln(bufT, P(20), P(19), bufX2, BB * LDEC);                       // ln1
    // cross attention: cross{bk7,bo8,bq9,bv10,wk11,wo12,wq13,wv14}
    attn_block(bufX2, encOut, LDEC, 512, P(13), P(9), P(11), P(7),
               P(14), P(10), P(12), P(8), 0, 0x51ED2709u, bufT);
    ln(bufT, P(22), P(21), bufX3, BB * LDEC);                       // ln2
    toBF(bufX3, bfA, (size_t)BB * LDEC * DM);
    wToBF(P(16), DM, DFF);
    gemmG(bfA, P(15), bufFF, BB * LDEC, DFF, DM);
    toBF(bufFF, bfA, (size_t)BB * LDEC * DFF);
    wToBF(P(18), DFF, DM);
    gemmR(bfA, P(17), bufX3, bufT, BB * LDEC, DM, DFF);
    ln(bufT, P(24), P(23), bufX, BB * LDEC);                        // ln3
    ln(bufX, P(34), P(33), bufX2, BB * LDEC);                       // dec_norm

    proj_kernel<<<(BB * PRED * 7 + 255) / 256, 256, 0, stream>>>(bufX2, P(77), P(76),
                                                                 (float*)d_out);
}
